// NNConvEmbedder_43439299231747
// MI455X (gfx1250) — matrix operand order repaired and checked by reading, not compiled, not run on hardware
//
#include <hip/hip_runtime.h>

typedef _Float16 v16h __attribute__((ext_vector_type(16)));
typedef _Float16 v8h  __attribute__((ext_vector_type(8)));
typedef float    v8f  __attribute__((ext_vector_type(8)));

#define HIDC 64
#define INN  32
#define OUTN 32
#define KTOT 2080          /* 64*32 + 32 (b2 folded into K) */
#define KROW 2088          /* padded LDS row stride in halves (bank-conflict-free) */
#define RH_STRIDE 72       /* padded rh tile row stride in halves */

// ---------------- zero accumulators ----------------
__global__ void kzero_kernel(float* aggr, int nAggr, float* pooled, int nPooled,
                             float* counts, int nCnt) {
  int i = blockIdx.x * blockDim.x + threadIdx.x;
  if (i < nAggr) { aggr[i] = 0.f; return; }
  int j = i - nAggr;
  if (j < nPooled) { pooled[j] = 0.f; return; }
  int k = j - nPooled;
  if (k < nCnt) counts[k] = 0.f;
}

// ---------------- pack W2' (n-major, padded, f16, b2 folded) ----------------
// w2t[row=o][col=k] ; k = h*32+i for k<2048 ; k-2048 = i for the b2 block
__global__ void k0_pack_w2t(const float* __restrict__ W2, const float* __restrict__ b2,
                            _Float16* __restrict__ w2t, int total) {
  int i = blockIdx.x * blockDim.x + threadIdx.x;
  if (i >= total) return;
  int row = i / KROW;
  int col = i % KROW;
  float v = 0.f;
  if (col < 2048) {
    int h  = col >> 5;
    int ii = col & 31;
    v = W2[h * 1024 + ii * 32 + row];
  } else if (col < KTOT) {
    int ii = col - 2048;
    v = b2[ii * 32 + row];
  }
  w2t[i] = (_Float16)v;
}

// ---------------- edge MLP: h = ea @ W1 + b1, block-partial stats ----------------
__global__ void k1_edge_mlp(const float* __restrict__ ea, const float* __restrict__ W1,
                            const float* __restrict__ b1, float* __restrict__ hraw,
                            float* __restrict__ pSum, float* __restrict__ pSq, int E) {
  extern __shared__ __align__(16) char dynsmem[];
  float* sEA  = (float*)dynsmem;        // 256*16
  float* sH   = sEA + 4096;             // 256*64
  float* sW1  = sH + 16384;             // 16*64
  float* sRed = sW1 + 1024;             // 2*256
  int t = threadIdx.x;
  int eBase = blockIdx.x * 256;
  for (int i = t; i < 1024; i += 256) sW1[i] = W1[i];
  for (int i = t; i < 4096; i += 256) {
    int e = eBase + (i >> 4);
    sEA[i] = (e < E) ? ea[(size_t)e * 16 + (i & 15)] : 0.f;
  }
  __syncthreads();
  int c = t & 63, q = t >> 6;
  float bc = b1[c];
  float s = 0.f, s2 = 0.f;
  for (int el = q * 64; el < q * 64 + 64; ++el) {
    float acc = bc;
#pragma unroll
    for (int i = 0; i < 16; ++i) acc = fmaf(sEA[el * 16 + i], sW1[i * 64 + c], acc);
    sH[el * 64 + c] = acc;
    if (eBase + el < E) { s += acc; s2 += acc * acc; }
  }
  sRed[q * 64 + c] = s;
  sRed[256 + q * 64 + c] = s2;
  __syncthreads();
  if (t < 64) {
    float a = 0.f, b = 0.f;
#pragma unroll
    for (int qq = 0; qq < 4; ++qq) { a += sRed[qq * 64 + t]; b += sRed[256 + qq * 64 + t]; }
    pSum[blockIdx.x * 64 + t] = a;
    pSq [blockIdx.x * 64 + t] = b;
  }
  __syncthreads();
  float4* dst = (float4*)(hraw + (size_t)eBase * 64);
  const float4* src = (const float4*)sH;
  for (int i = t; i < 4096; i += 256) {
    if (eBase + (i >> 4) < E) dst[i] = src[i];
  }
}

// ---------------- finalize BN (deterministic fixed-order reduction) ----------------
__global__ void k2_bn_finalize(const float* __restrict__ pSum, const float* __restrict__ pSq,
                               const float* __restrict__ gamma, const float* __restrict__ beta,
                               float* __restrict__ bnScale, float* __restrict__ bnShift,
                               int nParts, float invE) {
  int c = threadIdx.x;   // 64 threads
  float s = 0.f, s2 = 0.f;
  for (int b = 0; b < nParts; ++b) { s += pSum[b * 64 + c]; s2 += pSq[b * 64 + c]; }
  float mu  = s * invE;
  float var = s2 * invE - mu * mu;
  float sc  = gamma[c] * rsqrtf(var + 1e-5f);
  bnScale[c] = sc;
  bnShift[c] = beta[c] - mu * sc;
}

// ---------------- main fused WMMA kernel ----------------
// msgs = [relu(bn(h)) outer x_src] @ W2' (+ x_src @ B2 via folded K rows),
// scatter-added into aggr[dst].
__global__ void k4_wmma(const float* __restrict__ x, const int* __restrict__ ei,
                        const float* __restrict__ hraw, const float* __restrict__ bnScale,
                        const float* __restrict__ bnShift, const _Float16* __restrict__ w2tG,
                        float* __restrict__ aggr, int E, int nTiles) {
  extern __shared__ __align__(16) char dynsmem[];
  _Float16* w2t  = (_Float16*)dynsmem;                       // 32*KROW halves
  float* sScale  = (float*)(dynsmem + 32 * KROW * 2);        // 64
  float* sShift  = sScale + 64;                              // 64
  _Float16* rhT  = (_Float16*)(sShift + 64);                 // 8 waves * 16*RH_STRIDE

  // vectorized cooperative LDS fill of W2'
  {
    const uint4* s = (const uint4*)w2tG;
    uint4* d = (uint4*)w2t;
    const int n16 = (32 * KROW * 2) / 16;
    for (int i = threadIdx.x; i < n16; i += blockDim.x) d[i] = s[i];
  }
  if (threadIdx.x < 64) {
    sScale[threadIdx.x] = bnScale[threadIdx.x];
    sShift[threadIdx.x] = bnShift[threadIdx.x];
  }
  __syncthreads();

  const int lane = threadIdx.x & 31;
  const int wave = threadIdx.x >> 5;
  const int hi   = lane >> 4;       // lane half
  const int m    = lane & 15;       // A/D row this lane stages, B/D column
  const int klow = hi ? 8 : 0;      // A-fragment K offset per 16-bit layout
  _Float16* myRh = rhT + wave * (16 * RH_STRIDE);
  const int stride = gridDim.x * 8;

  for (int tile = blockIdx.x * 8 + wave; tile < nTiles; tile += stride) {
    int e = tile * 16 + m;
    bool valid = (e < E);
    int ec = valid ? e : (E - 1);
    int srcN = ei[ec];
    int dstN = ei[E + ec];
    float msk = valid ? 1.f : 0.f;

    // fixed per-tile x_src fragment (A layout: lanes<16 K{0..7,16..23}, lanes>=16 K{8..15,24..31})
    const float* xr = x + (size_t)srcN * 32;
    v16h ax;
#pragma unroll
    for (int t2 = 0; t2 < 8; ++t2) ax[t2] = (_Float16)(msk * xr[klow + t2]);
#pragma unroll
    for (int t2 = 0; t2 < 8; ++t2) ax[8 + t2] = (_Float16)(msk * xr[16 + klow + t2]);

    // stage BN+ReLU'd h row into this wave's private LDS tile (fused BN)
    {
      const int chalf = hi * 32;
      const float* hr = hraw + (size_t)ec * 64 + chalf;
#pragma unroll
      for (int cc = 0; cc < 32; ++cc) {
        float v = fmaf(hr[cc], sScale[chalf + cc], sShift[chalf + cc]);
        v = v > 0.f ? v : 0.f;
        myRh[m * RH_STRIDE + chalf + cc] = (_Float16)v;
      }
    }
    __builtin_amdgcn_wave_barrier();

    // prefetch next tile's h rows
    if (tile + stride < nTiles) {
      int en = (tile + stride) * 16 + m;
      if (en < E) __builtin_prefetch(hraw + (size_t)en * 64, 0, 1);
    }

    v8f c0 = {}; v8f c1 = {};
    const _Float16* brow0 = w2t + (size_t)m        * KROW + (hi << 3);
    const _Float16* brow1 = w2t + (size_t)(m + 16) * KROW + (hi << 3);
    for (int j = 0; j < 65; ++j) {
      // chunk j of A is rank-1: rh[e_m, j] * x_src ; last chunk is the b2 block (scale 1)
      _Float16 rh = (j < 64) ? myRh[m * RH_STRIDE + j] : (_Float16)1.0f;
      v16h a = ax * rh;                                   // v_pk_mul_f16 x8
      const _Float16* p0 = brow0 + j * 32;
      const _Float16* p1 = brow1 + j * 32;
      v8h b0l = *(const v8h*)p0, b0h = *(const v8h*)(p0 + 16);   // ds_load_b128 pairs
      v8h b1l = *(const v8h*)p1, b1h = *(const v8h*)(p1 + 16);
      v16h bb0 = __builtin_shufflevector(b0l, b0h, 0,1,2,3,4,5,6,7,8,9,10,11,12,13,14,15);
      v16h bb1 = __builtin_shufflevector(b1l, b1h, 0,1,2,3,4,5,6,7,8,9,10,11,12,13,14,15);
      c0 = __builtin_amdgcn_wmma_f32_16x16x32_f16(false, a, false, bb0, (short)0, c0, false, false);
      c1 = __builtin_amdgcn_wmma_f32_16x16x32_f16(false, a, false, bb1, (short)0, c1, false, false);
    }

    // D layout: lane half selects M block (r -> M = 8*hi + r), N = m (+16 for tile 1)
#pragma unroll
    for (int r = 0; r < 8; ++r) {
      int mm = (hi << 3) + r;
      int dd = __shfl(dstN, mm, 32);
      atomicAdd(&aggr[(size_t)dd * 32 + m],      c0[r]);
      atomicAdd(&aggr[(size_t)dd * 32 + 16 + m], c1[r]);
    }
    __builtin_amdgcn_wave_barrier();
  }
}

// ---------------- root transform + graph mean-pool accumulate ----------------
__global__ void k5_root_pool(const float* __restrict__ x, const float* __restrict__ Wr,
                             const float* __restrict__ bias, const int* __restrict__ batch,
                             const float* __restrict__ aggr, float* __restrict__ pooled,
                             float* __restrict__ counts, int N) {
  __shared__ float sWr[INN * OUTN];
  __shared__ float sB[OUTN];
  for (int i = threadIdx.x; i < INN * OUTN; i += blockDim.x) sWr[i] = Wr[i];
  if (threadIdx.x < OUTN) sB[threadIdx.x] = bias[threadIdx.x];
  __syncthreads();
  int idx = blockIdx.x * blockDim.x + threadIdx.x;
  if (idx >= N * OUTN) return;
  int n = idx >> 5, o = idx & 31;
  const float* xr = x + (size_t)n * INN;
  float acc = aggr[idx] + sB[o];
#pragma unroll
  for (int i = 0; i < INN; ++i) acc = fmaf(xr[i], sWr[i * OUTN + o], acc);
  int g = batch[n];
  atomicAdd(&pooled[(size_t)g * OUTN + o], acc);
  if (o == 0) atomicAdd(&counts[g], 1.f);
}

// ---------------- mean divide ----------------
__global__ void k6_div(const float* __restrict__ pooled, const float* __restrict__ counts,
                       float* __restrict__ out, int total) {
  int i = blockIdx.x * blockDim.x + threadIdx.x;
  if (i < total) out[i] = pooled[i] / fmaxf(counts[i >> 5], 1.f);
}

extern "C" void kernel_launch(void* const* d_in, const int* in_sizes, int n_in,
                              void* d_out, int out_size, void* d_ws, size_t ws_size,
                              hipStream_t stream) {
  const float* x     = (const float*)d_in[0];
  const float* ea    = (const float*)d_in[1];
  const int*   ei    = (const int*)d_in[2];
  const int*   batch = (const int*)d_in[3];
  const float* W1    = (const float*)d_in[4];
  const float* b1    = (const float*)d_in[5];
  const float* gamma = (const float*)d_in[6];
  const float* beta  = (const float*)d_in[7];
  const float* W2    = (const float*)d_in[8];
  const float* b2    = (const float*)d_in[9];
  const float* Wr    = (const float*)d_in[10];
  const float* bias  = (const float*)d_in[11];

  const int N = in_sizes[0] / INN;        // 10000
  const int E = in_sizes[1] / 16;         // 320000
  const int G = out_size / OUTN;          // 256
  const int nParts = (E + 255) / 256;
  const int nTiles = (E + 15) / 16;

  // workspace layout (256B aligned slabs)
  size_t off = 0;
  auto alloc = [&](size_t bytes) -> size_t {
    size_t cur = off;
    off = (off + bytes + 255) & ~(size_t)255;
    return cur;
  };
  char* ws = (char*)d_ws;
  float*    hraw    = (float*)(ws + alloc((size_t)E * 64 * sizeof(float)));
  float*    pSum    = (float*)(ws + alloc((size_t)nParts * 64 * sizeof(float)));
  float*    pSq     = (float*)(ws + alloc((size_t)nParts * 64 * sizeof(float)));
  float*    bnScale = (float*)(ws + alloc(64 * sizeof(float)));
  float*    bnShift = (float*)(ws + alloc(64 * sizeof(float)));
  _Float16* w2t     = (_Float16*)(ws + alloc((size_t)32 * KROW * sizeof(_Float16)));
  float*    aggr    = (float*)(ws + alloc((size_t)N * OUTN * sizeof(float)));
  float*    pooled  = (float*)(ws + alloc((size_t)G * OUTN * sizeof(float)));
  float*    counts  = (float*)(ws + alloc((size_t)G * sizeof(float)));

  // 1) zero accumulators
  {
    int total = N * OUTN + G * OUTN + G;
    kzero_kernel<<<(total + 255) / 256, 256, 0, stream>>>(aggr, N * OUTN, pooled, G * OUTN, counts, G);
  }
  // 2) pack W2' (f16, padded, b2 folded)
  {
    int total = 32 * KROW;
    k0_pack_w2t<<<(total + 255) / 256, 256, 0, stream>>>(W2, b2, w2t, total);
  }
  // 3) edge MLP + partial stats
  {
    size_t smem1 = (4096 + 16384 + 1024 + 512) * sizeof(float);  // ~86KB
    k1_edge_mlp<<<nParts, 256, smem1, stream>>>(ea, W1, b1, hraw, pSum, pSq, E);
  }
  // 4) BN finalize
  k2_bn_finalize<<<1, 64, 0, stream>>>(pSum, pSq, gamma, beta, bnScale, bnShift, nParts, 1.0f / (float)E);
  // 5) main WMMA GEMM + scatter
  {
    size_t smem4 = (size_t)32 * KROW * 2 + 128 * sizeof(float) + (size_t)8 * 16 * RH_STRIDE * 2; // ~149KB
    k4_wmma<<<512, 256, smem4, stream>>>(x, ei, hraw, bnScale, bnShift, w2t, aggr, E, nTiles);
  }
  // 6) root transform + pooling accumulate
  k5_root_pool<<<(N * OUTN + 255) / 256, 256, 0, stream>>>(x, Wr, bias, batch, aggr, pooled, counts, N);
  // 7) mean divide into d_out
  k6_div<<<(G * OUTN + 255) / 256, 256, 0, stream>>>(pooled, counts, (float*)d_out, G * OUTN);
}
